// GAT_58119497450254
// MI455X (gfx1250) — matrix operand order repaired and checked by reading, not compiled, not run on hardware
//
#include <hip/hip_runtime.h>

#define NN 20000
#define EE 320000
#define EPE 340000   // E + N self loops
#define GG 64
#define HH 4
#define CC 64
#define HIDC 256
#define IN_DIM 128
#define LATC 64
#define NEGS 0.2f
#define BN_EPS 1e-5f

typedef __attribute__((ext_vector_type(16))) _Float16 v16h;
typedef __attribute__((ext_vector_type(8)))  _Float16 v8h;
typedef __attribute__((ext_vector_type(8)))  float    v8f;

__device__ __forceinline__ void atomicMaxF32(float* addr, float val) {
  // monotone int mapping: works with -inf init, mixed signs
  if (val >= 0.0f) atomicMax((int*)addr, __float_as_int(val));
  else             atomicMin((unsigned int*)addr, __float_as_uint(val));
}

// ---------------- f32 -> f16 convert ----------------
__global__ void k_cvt_f16(const float* __restrict__ src, _Float16* __restrict__ dst, int n) {
  int i = blockIdx.x * blockDim.x + threadIdx.x;
  if (i < n) dst[i] = (_Float16)src[i];
}

// ---------------- W (K x 256, f32) -> wT (256 x K, f16) ----------------
__global__ void k_cvt_wT(const float* __restrict__ W, _Float16* __restrict__ wT, int K) {
  int i = blockIdx.x * blockDim.x + threadIdx.x;
  if (i >= K * HIDC) return;
  int k = i / HIDC, n = i % HIDC;
  wT[n * K + k] = (_Float16)W[k * HIDC + n];
}

// ---------------- WMMA GEMM: C[M,256] = A[M,K] * B[K,256] ----------------
// A: f16 row-major (M x K).  B given transposed: wT[256 x K] f16 row-major.
// One 16x16 output tile per wave. 8 waves / block of 256 threads.
template <int K>
__global__ void __launch_bounds__(256)
k_gemm_wmma(const _Float16* __restrict__ A,
            const _Float16* __restrict__ wT,
            float* __restrict__ Cmat) {
  const int TN = HIDC / 16;                  // 16 N-tiles
  int wave = threadIdx.x >> 5;
  int lane = threadIdx.x & 31;
  int tile = blockIdx.x * 8 + wave;
  int mt = tile / TN, nt = tile % TN;
  if (mt * 16 >= NN) return;                 // wave-uniform guard (EXEC stays all-1)
  int half = lane >> 4;                      // 0: lanes 0-15, 1: lanes 16-31
  int r    = lane & 15;
  int mrow = mt * 16 + r;                    // A row for this lane
  int ncol = nt * 16 + r;                    // B column for this lane
  const _Float16* arow = A  + (size_t)mrow * K;
  const _Float16* bcol = wT + (size_t)ncol * K;
  v8f c = {};
#pragma unroll
  for (int kb = 0; kb < K; kb += 32) {
    // A 16-bit 16x32 layout: lanes0-15 hold K = {0..7, 16..23}, lanes16-31 K = {8..15, 24..31}
    v8h alo = *(const v8h*)(arow + kb + half * 8);
    v8h ahi = *(const v8h*)(arow + kb + 16 + half * 8);
    v16h a;
#pragma unroll
    for (int j = 0; j < 8; ++j) { a[j] = alo[j]; a[j + 8] = ahi[j]; }
    // B 32x16 layout: lanes0-15 hold K = {0..7, 16..23} (col = lane), lanes16-31 K = {8..15, 24..31}
    v8h blo = *(const v8h*)(bcol + kb + half * 8);
    v8h bhi = *(const v8h*)(bcol + kb + 16 + half * 8);
    v16h b;
#pragma unroll
    for (int j = 0; j < 8; ++j) { b[j] = blo[j]; b[j + 8] = bhi[j]; }
    c = __builtin_amdgcn_wmma_f32_16x16x32_f16(false, a, false, b, (short)0, c, false, false);
  }
  // C/D f32 16x16: VGPR rr -> row (rr + half*8), col = lane&15
#pragma unroll
  for (int rr = 0; rr < 8; ++rr) {
    int mo = mt * 16 + rr + half * 8;
    Cmat[(size_t)mo * HIDC + nt * 16 + r] = c[rr];
  }
}

// ---------------- per-(node,head) attention scalars ----------------
__global__ void k_attn(const float* __restrict__ xp, const float* __restrict__ a_s,
                       const float* __restrict__ a_d, float* __restrict__ als,
                       float* __restrict__ ald) {
  int i = blockIdx.x * blockDim.x + threadIdx.x;
  if (i >= NN * HH) return;
  int n = i >> 2, h = i & 3;
  const float* v  = xp + (size_t)n * HIDC + h * CC;
  const float* s1 = a_s + h * CC;
  const float* s2 = a_d + h * CC;
  float s = 0.0f, d = 0.0f;
#pragma unroll 8
  for (int c = 0; c < CC; ++c) { float x = v[c]; s += x * s1[c]; d += x * s2[c]; }
  als[i] = s; ald[i] = d;
}

// ---------------- per-layer re-init: acc=0, emax=-inf, denom=0 ----------------
__global__ void k_init_layer(float* __restrict__ acc, float* __restrict__ emax,
                             float* __restrict__ denom) {
  int i = blockIdx.x * blockDim.x + threadIdx.x;
  if (i < NN * HIDC) acc[i] = 0.0f;
  if (i < NN * HH) { emax[i] = -__builtin_inff(); denom[i] = 0.0f; }
}

// ---------------- edge pass 1: logit + segment max ----------------
__global__ void k_edge_logit(const int* __restrict__ ei, const float* __restrict__ als,
                             const float* __restrict__ ald, float* __restrict__ ee,
                             float* __restrict__ emax) {
  int i = blockIdx.x * blockDim.x + threadIdx.x;
  if (i >= EPE * HH) return;
  int ed = i >> 2, h = i & 3;
  int s, d;
  if (ed < EE) { s = ei[ed]; d = ei[EE + ed]; } else { s = ed - EE; d = s; }
  float v = als[s * HH + h] + ald[d * HH + h];
  v = v > 0.0f ? v : NEGS * v;
  ee[i] = v;
  atomicMaxF32(&emax[d * HH + h], v);
}

// ---------------- edge pass 2: exp + segment sum ----------------
__global__ void k_edge_exp(const int* __restrict__ ei, float* __restrict__ ee,
                           const float* __restrict__ emax, float* __restrict__ denom) {
  int i = blockIdx.x * blockDim.x + threadIdx.x;
  if (i >= EPE * HH) return;
  int ed = i >> 2, h = i & 3;
  int d = (ed < EE) ? ei[EE + ed] : (ed - EE);
  float ex = __expf(ee[i] - emax[d * HH + h]);
  ee[i] = ex;
  atomicAdd(&denom[d * HH + h], ex);
}

// ---------------- edge pass 3: weighted scatter-add (4 channels / thread) ----------------
__global__ void k_edge_scatter(const int* __restrict__ ei, const float* __restrict__ ee,
                               const float* __restrict__ denom, const float* __restrict__ xp,
                               float* __restrict__ acc) {
  int i = blockIdx.x * blockDim.x + threadIdx.x;
  if (i >= EPE * (HIDC / 4)) return;
  int ed = i >> 6;               // 64 groups of 4 channels
  int c4 = (i & 63) * 4;         // channel base 0..252
  int h  = c4 >> 6;              // head = channel/64
  int s, d;
  if (ed < EE) { s = ei[ed]; d = ei[EE + ed]; } else { s = ed - EE; d = s; }
  float alpha = ee[ed * HH + h] / (denom[d * HH + h] + 1e-16f);
  const float4 xv = *(const float4*)(xp + (size_t)s * HIDC + c4);
  float* dst = acc + (size_t)d * HIDC + c4;
  atomicAdd(dst + 0, alpha * xv.x);
  atomicAdd(dst + 1, alpha * xv.y);
  atomicAdd(dst + 2, alpha * xv.z);
  atomicAdd(dst + 3, alpha * xv.w);
}

// ---------------- bias + BN(eval) + ReLU ----------------
__global__ void k_epilogue(const float* __restrict__ acc, const float* __restrict__ bias,
                           const float* __restrict__ g, const float* __restrict__ b,
                           const float* __restrict__ m, const float* __restrict__ v,
                           float* __restrict__ hout) {
  int i = blockIdx.x * blockDim.x + threadIdx.x;
  if (i >= NN * HIDC) return;
  int c = i & (HIDC - 1);
  float x = acc[i] + bias[c];
  x = (x - m[c]) * rsqrtf(v[c] + BN_EPS) * g[c] + b[c];
  hout[i] = x > 0.0f ? x : 0.0f;
}

// ---------------- pooling ----------------
__global__ void k_pool_init(float* __restrict__ pooled) {
  int i = blockIdx.x * blockDim.x + threadIdx.x;
  if (i < GG * HIDC) pooled[i] = 0.0f;
}
__global__ void k_pool(const float* __restrict__ h, const int* __restrict__ batch,
                       float* __restrict__ pooled) {
  int i = blockIdx.x * blockDim.x + threadIdx.x;
  if (i >= NN * HIDC) return;
  int n = i >> 8, c = i & (HIDC - 1);
  atomicAdd(&pooled[batch[n] * HIDC + c], h[i]);
}

// ---------------- final BN + FC ----------------
__global__ void k_final(const float* __restrict__ pooled, const float* __restrict__ lg,
                        const float* __restrict__ lb, const float* __restrict__ lm,
                        const float* __restrict__ lv, const float* __restrict__ fw,
                        const float* __restrict__ fb, float* __restrict__ out) {
  int i = blockIdx.x * blockDim.x + threadIdx.x;
  if (i >= GG * LATC) return;
  int g = i / LATC, l = i % LATC;
  float s = 0.0f;
#pragma unroll 4
  for (int k = 0; k < HIDC; ++k) {
    float x = (pooled[g * HIDC + k] - lm[k]) * rsqrtf(lv[k] + BN_EPS) * lg[k] + lb[k];
    s += x * fw[k * LATC + l];
  }
  out[i] = s + fb[l];
}

extern "C" void kernel_launch(void* const* d_in, const int* in_sizes, int n_in,
                              void* d_out, int out_size, void* d_ws, size_t ws_size,
                              hipStream_t stream) {
  (void)in_sizes; (void)n_in; (void)out_size; (void)ws_size;
  const float* x    = (const float*)d_in[0];
  const int*   ei   = (const int*)d_in[1];
  const int*   batch= (const int*)d_in[2];
  const float* W0   = (const float*)d_in[3];
  const float* Wr   = (const float*)d_in[4];
  const float* asrc = (const float*)d_in[5];
  const float* adst = (const float*)d_in[6];
  const float* bconv= (const float*)d_in[7];
  const float* bng  = (const float*)d_in[8];
  const float* bnb  = (const float*)d_in[9];
  const float* bnm  = (const float*)d_in[10];
  const float* bnv  = (const float*)d_in[11];
  const float* lg   = (const float*)d_in[12];
  const float* lb   = (const float*)d_in[13];
  const float* lm   = (const float*)d_in[14];
  const float* lv   = (const float*)d_in[15];
  const float* fw   = (const float*)d_in[16];
  const float* fb   = (const float*)d_in[17];
  float* out = (float*)d_out;

  // carve workspace (256B-aligned blocks), ~79 MB total
  char* p = (char*)d_ws;
  auto carve = [&](size_t bytes) -> char* {
    char* q = p; p += (bytes + 255) & ~(size_t)255; return q;
  };
  float*    xp    = (float*)   carve((size_t)NN * HIDC * 4);
  float*    accb  = (float*)   carve((size_t)NN * HIDC * 4);
  float*    hf32  = (float*)   carve((size_t)NN * HIDC * 4);
  _Float16* hf16  = (_Float16*)carve((size_t)NN * HIDC * 2);
  _Float16* wT    = (_Float16*)carve((size_t)HIDC * HIDC * 2);
  float*    als   = (float*)   carve((size_t)NN * HH * 4);
  float*    ald   = (float*)   carve((size_t)NN * HH * 4);
  float*    emax  = (float*)   carve((size_t)NN * HH * 4);
  float*    denom = (float*)   carve((size_t)NN * HH * 4);
  float*    ee    = (float*)   carve((size_t)EPE * HH * 4);
  float*    pooled= (float*)   carve((size_t)GG * HIDC * 4);

  const int T = 256;
  const int gemm_blocks = (NN / 16) * (HIDC / 16) / 8;   // 2500 blocks, 8 waves each

  for (int i = 0; i < 3; ++i) {
    const int K = (i == 0) ? IN_DIM : HIDC;
    const float* Asrc = (i == 0) ? x : hf32;
    const float* Wsrc = (i == 0) ? W0 : (Wr + (size_t)(i - 1) * HIDC * HIDC);
    const int nA = NN * K;

    k_cvt_f16<<<(nA + T - 1) / T, T, 0, stream>>>(Asrc, hf16, nA);
    k_cvt_wT<<<(K * HIDC + T - 1) / T, T, 0, stream>>>(Wsrc, wT, K);
    if (i == 0) k_gemm_wmma<IN_DIM><<<gemm_blocks, T, 0, stream>>>(hf16, wT, xp);
    else        k_gemm_wmma<HIDC>  <<<gemm_blocks, T, 0, stream>>>(hf16, wT, xp);

    k_attn<<<(NN * HH + T - 1) / T, T, 0, stream>>>(xp, asrc + i * HH * CC,
                                                    adst + i * HH * CC, als, ald);
    k_init_layer<<<(NN * HIDC + T - 1) / T, T, 0, stream>>>(accb, emax, denom);
    k_edge_logit<<<(EPE * HH + T - 1) / T, T, 0, stream>>>(ei, als, ald, ee, emax);
    k_edge_exp<<<(EPE * HH + T - 1) / T, T, 0, stream>>>(ei, ee, emax, denom);
    k_edge_scatter<<<(EPE * (HIDC / 4) + T - 1) / T, T, 0, stream>>>(ei, ee, denom, xp, accb);
    k_epilogue<<<(NN * HIDC + T - 1) / T, T, 0, stream>>>(accb, bconv + i * HIDC,
                                                          bng + i * HIDC, bnb + i * HIDC,
                                                          bnm + i * HIDC, bnv + i * HIDC, hf32);
  }

  k_pool_init<<<(GG * HIDC + T - 1) / T, T, 0, stream>>>(pooled);
  k_pool<<<(NN * HIDC + T - 1) / T, T, 0, stream>>>(hf32, batch, pooled);
  k_final<<<(GG * LATC + T - 1) / T, T, 0, stream>>>(pooled, lg, lb, lm, lv, fw, fb, out);
}
